// VQVAE_47751446397407
// MI455X (gfx1250) — matrix operand order repaired and checked by reading, not compiled, not run on hardware
//
#include <hip/hip_runtime.h>
#include <hip/hip_bf16.h>

typedef __attribute__((ext_vector_type(16))) _Float16 v16h;
typedef __attribute__((ext_vector_type(8)))  _Float16 v8h;
typedef __attribute__((ext_vector_type(8)))  float    v8f;
typedef __attribute__((ext_vector_type(4)))  float    v4f;

// ---------------------------------------------------------------------------
// A-fragment loader: 16-bit A 16x32, contiguous K order (k = 16h + s).
// lanes 0-15 : row M=lane,    K 0..15
// lanes 16-31: row M=lane-16, K 16..31
// sA is row-major [128][32] f16 in LDS.
// ---------------------------------------------------------------------------
__device__ inline v16h lds_afrag(const _Float16* sA, int mbase, int lane) {
  const int l15 = lane & 15, hi = lane >> 4;
  const _Float16* p = sA + (mbase + l15) * 32 + hi * 16;
  v8h lo = *(const v8h*)p;         // K 0..7   or 16..23
  v8h hh = *(const v8h*)(p + 8);   // K 8..15  or 24..31
  v16h r;
#pragma unroll
  for (int i = 0; i < 8; ++i) { r[i] = lo[i]; r[i + 8] = hh[i]; }
  return r;
}

// ---------------------------------------------------------------------------
// Generic implicit-GEMM conv engine.
//   C[M x N] = A[M x K] * B[K x N]; A gathered (im2col) by Map::fetchRow
//   (16 consecutive K for one M row, branchless); B pre-packed f16 as W[n][k].
//   256 threads = 8 waves; wave w computes rows [m_tile+16w,+16) x NTILES*16.
//   KTOT multiple of 32, N = 16*NTILES.
// ---------------------------------------------------------------------------
template<int KTOT, int NTILES, class Map>
__global__ __launch_bounds__(256) void gemm_conv(Map map, const _Float16* __restrict__ Wbase) {
  __shared__ _Float16 sA[128 * 32];   // 8 KB
  const int tid = threadIdx.x, lane = tid & 31, wid = tid >> 5;
  const int m_tile = blockIdx.x * 128;
  const _Float16* W = map.selectW(m_tile, Wbase);
  const int mrowA = m_tile + (tid >> 1);     // staging row for this thread
  const int koffA = (tid & 1) << 4;          // 0 or 16 within the 32-K slab

  v8f zero = {};
  v8f acc[NTILES];
#pragma unroll
  for (int i = 0; i < NTILES; ++i) acc[i] = zero;

  for (int kc = 0; kc < KTOT; kc += 32) {
    // issue global gathers early (overlaps other waves' WMMA on previous tile)
    _Float16 row[16];
    map.fetchRow(mrowA, kc + koffA, row);
    __syncthreads();                         // previous sA fully consumed
    v8h a0, a1;
#pragma unroll
    for (int i = 0; i < 8; ++i) { a0[i] = row[i]; a1[i] = row[i + 8]; }
    v8h* dsp = (v8h*)(sA + tid * 16);        // 32B-aligned -> 2x ds_store_b128
    dsp[0] = a0; dsp[1] = a1;
    __syncthreads();

    v16h a = lds_afrag(sA, wid * 16, lane);
    if (kc + 32 < KTOT)                      // global_prefetch_b8 of next B slab
      __builtin_prefetch(W + (lane & 15) * KTOT + kc + 32, 0, 1);
#pragma unroll
    for (int nt = 0; nt < NTILES; ++nt) {
      // B 32x16: lane n<16 -> col n, K 0..15; lane>=16 -> col n-16, K 16..31
      v16h b = *(const v16h*)(W + (nt * 16 + (lane & 15)) * KTOT + kc + (lane >> 4) * 16);
      acc[nt] = __builtin_amdgcn_wmma_f32_16x16x32_f16(
          false, a, false, b, (short)0, acc[nt], false, false);
    }
  }
  // C layout: VGPR r = (M = r + 8*(lane>=16), N = lane&15); rows pixel-contig.
  const int hi = lane >> 4, l15 = lane & 15;
  const int mrow = m_tile + wid * 16 + hi * 8;
#pragma unroll
  for (int nt = 0; nt < NTILES; ++nt) {
    float v[8];
#pragma unroll
    for (int r = 0; r < 8; ++r) v[r] = acc[nt][r];
    map.storeRow(mrow, nt * 16 + l15, v);
  }
}

// ---------------------------------------------------------------------------
// Mappers: branchless im2col row gathers + vectorized epilogues
// ---------------------------------------------------------------------------
struct MapConv1 {                 // 3->128, k4 s2 p1, in 32x32 -> out 16x16
  const float* x; const float* bias; _Float16* h1;
  __device__ const _Float16* selectW(int, const _Float16* W) const { return W; }
  __device__ void fetchRow(int m, int kbase, _Float16* dst) const {
    int b = m >> 8, pix = m & 255, oy = pix >> 4, ox = pix & 15;
    int cin = kbase >> 4;                    // kbase 16-aligned: one cin per row
    bool cv = cin < 3;                       // K padded 48->64
    const float* base = x + (size_t)(b * 3 + (cv ? cin : 0)) * 1024;
#pragma unroll
    for (int ky = 0; ky < 4; ++ky) {
      int iy = oy * 2 - 1 + ky;
      bool vy = cv & ((unsigned)iy < 32u);
#pragma unroll
      for (int kx = 0; kx < 4; ++kx) {
        int ix = ox * 2 - 1 + kx;
        bool vv = vy & ((unsigned)ix < 32u);
        float t = base[vv ? iy * 32 + ix : 0];
        dst[ky * 4 + kx] = (_Float16)(vv ? t : 0.f);
      }
    }
  }
  __device__ void storeRow(int mrow, int n, const float* v) const {
    int b = mrow >> 8, pix = mrow & 255;     // 8 consecutive pixels
    float bb = bias[n];
    v8h o;
#pragma unroll
    for (int r = 0; r < 8; ++r) { float t = v[r] + bb; o[r] = (_Float16)(t > 0.f ? t : 0.f); }
    *(v8h*)(h1 + ((b * 128 + n) << 8) + pix) = o;
  }
};

struct MapConv2 {                 // 128->128, k4 s2 p1, 16x16 -> 8x8
  const _Float16* h1; const float* bias; _Float16* h2;
  __device__ const _Float16* selectW(int, const _Float16* W) const { return W; }
  __device__ void fetchRow(int m, int kbase, _Float16* dst) const {
    int b = m >> 6, pix = m & 63, oy = pix >> 3, ox = pix & 7;
    int cin = kbase >> 4;
    const _Float16* base = h1 + ((b * 128 + cin) << 8);
#pragma unroll
    for (int ky = 0; ky < 4; ++ky) {
      int iy = oy * 2 - 1 + ky;
      bool vy = (unsigned)iy < 16u;
#pragma unroll
      for (int kx = 0; kx < 4; ++kx) {
        int ix = ox * 2 - 1 + kx;
        bool vv = vy & ((unsigned)ix < 16u);
        _Float16 t = base[vv ? (iy << 4) + ix : 0];
        dst[ky * 4 + kx] = vv ? t : (_Float16)0.f;
      }
    }
  }
  __device__ void storeRow(int mrow, int n, const float* v) const {
    int b = mrow >> 6, pix = mrow & 63;
    float bb = bias[n];
    v8h o;
#pragma unroll
    for (int r = 0; r < 8; ++r) { float t = v[r] + bb; o[r] = (_Float16)(t > 0.f ? t : 0.f); }
    *(v8h*)(h2 + ((b * 128 + n) << 6) + pix) = o;
  }
};

struct MapConv3 {                 // 128->64, k3 s1 p1, 8x8 (bias added in pool)
  const _Float16* h2; float* c3;
  __device__ const _Float16* selectW(int, const _Float16* W) const { return W; }
  __device__ void fetchRow(int m, int kbase, _Float16* dst) const {
    int b = m >> 6, pix = m & 63, oy = pix >> 3, ox = pix & 7;
    const _Float16* hb = h2 + ((size_t)b << 13);   // b*128*64
#pragma unroll
    for (int e = 0; e < 16; ++e) {
      int k = kbase + e;
      int cin = k / 9, tap = k - cin * 9, ky = tap / 3, kx = tap - ky * 3;
      int iy = oy - 1 + ky, ix = ox - 1 + kx;
      bool vv = ((unsigned)iy < 8u) & ((unsigned)ix < 8u);
      _Float16 t = hb[(cin << 6) + (vv ? (iy << 3) + ix : 0)];
      dst[e] = vv ? t : (_Float16)0.f;
    }
  }
  __device__ void storeRow(int mrow, int n, const float* v) const {
    int b = mrow >> 6, pix = mrow & 63;
    float* base = c3 + ((b * 64 + n) << 6) + pix;
    v4f o0, o1;
#pragma unroll
    for (int r = 0; r < 4; ++r) { o0[r] = v[r]; o1[r] = v[r + 4]; }
    *(v4f*)base = o0; *(v4f*)(base + 4) = o1;
  }
};

struct MapH1 {                    // decoder convT1 collapsed to 4x4 type grid
  const _Float16* qf; const float* bias; _Float16* h1t;
  __device__ const _Float16* selectW(int, const _Float16* W) const {
    return W + (size_t)blockIdx.y * (128 * 64);            // per (rt,ct) matrix
  }
  __device__ void fetchRow(int m, int kbase, _Float16* dst) const {
    const v8h* s = (const v8h*)(qf + (m << 6) + kbase);    // contiguous, aligned
    v8h a = s[0], b = s[1];
#pragma unroll
    for (int i = 0; i < 8; ++i) { dst[i] = a[i]; dst[i + 8] = b[i]; }
  }
  __device__ void storeRow(int mrow, int n, const float* v) const {
    float bb = bias[n];
#pragma unroll
    for (int r = 0; r < 8; ++r) {
      float t = v[r] + bb; t = t > 0.f ? t : 0.f;
      h1t[(((mrow + r) << 4) + blockIdx.y) * 128 + n] = (_Float16)t;
    }
  }
};

struct MapT2 {                    // convT2 128->128, k4 s2 p1, 16x16 -> 32x32
  const _Float16* h1t; const float* bias; _Float16* h2d; int b0;
  __device__ const _Float16* selectW(int m_tile, const _Float16* W) const {
    return W + (size_t)((m_tile >> 8) & 3) * (128 * 512);  // parity-class pack
  }
  __device__ void fetchRow(int m, int kbase, _Float16* dst) const {
    int bl = m >> 10, r = m & 1023;
    int pclass = r >> 8, py = pclass >> 1, px = pclass & 1;
    int r2 = r & 255;
    int oy = ((r2 >> 4) << 1) + py, ox = ((r2 & 15) << 1) + px;
    int b = b0 + bl;
    int  tb[4]; bool tv[4];                  // per-ts type base, computed once
#pragma unroll
    for (int ts = 0; ts < 4; ++ts) {
      int ky = (1 - py) + ((ts >> 1) << 1), kx = (1 - px) + ((ts & 1) << 1);
      int ty = oy + 1 - ky, tx = ox + 1 - kx;
      int iy = ty >> 1, ix = tx >> 1;
      bool v = (ty >= 0) & (tx >= 0) & (iy < 16) & (ix < 16);
      int rt = (iy == 0) ? 0 : (iy == 15) ? 3 : ((iy & 1) ? 2 : 1);
      int ct = (ix == 0) ? 0 : (ix == 15) ? 3 : ((ix & 1) ? 2 : 1);
      tv[ts] = v;
      tb[ts] = v ? ((b << 4) + rt * 4 + ct) * 128 : 0;
    }
    int cin0 = kbase >> 2;
#pragma unroll
    for (int e = 0; e < 16; ++e) {
      int cin = cin0 + (e >> 2), ts = e & 3;
      _Float16 t = h1t[tb[ts] + cin];
      dst[e] = tv[ts] ? t : (_Float16)0.f;
    }
  }
  __device__ void storeRow(int mrow, int n, const float* v) const {
    int bl = mrow >> 10, r = mrow & 1023;
    int pclass = r >> 8, py = pclass >> 1, px = pclass & 1;
    int r2 = r & 255;
    int oy = ((r2 >> 4) << 1) + py;          // constant over the 8-row run
    int oxb = ((r2 & 15) << 1) + px;         // ox steps by 2 per row
    _Float16* base = h2d + (bl * 128 + n) * 1024 + oy * 32;
    float bb = bias[n];
#pragma unroll
    for (int rr = 0; rr < 8; ++rr) {
      float t = v[rr] + bb; t = t > 0.f ? t : 0.f;
      base[oxb + rr * 2] = (_Float16)t;
    }
  }
};

struct MapD3 {                    // 128->3 (N padded 16), k3 s1 p1, 32x32
  const _Float16* h2d; const float* bias; float* out; int b0;
  __device__ const _Float16* selectW(int, const _Float16* W) const { return W; }
  __device__ void fetchRow(int m, int kbase, _Float16* dst) const {
    int bl = m >> 10, pix = m & 1023, oy = pix >> 5, ox = pix & 31;
    const _Float16* hb = h2d + ((size_t)bl << 17);  // bl*128*1024
#pragma unroll
    for (int e = 0; e < 16; ++e) {
      int k = kbase + e;
      int cin = k / 9, tap = k - cin * 9, ky = tap / 3, kx = tap - ky * 3;
      int iy = oy - 1 + ky, ix = ox - 1 + kx;
      bool vv = ((unsigned)iy < 32u) & ((unsigned)ix < 32u);
      _Float16 t = hb[(cin << 10) + (vv ? (iy << 5) + ix : 0)];
      dst[e] = vv ? t : (_Float16)0.f;
    }
  }
  __device__ void storeRow(int mrow, int n, const float* v) const {
    if (n >= 3) return;
    int bl = mrow >> 10, pix = mrow & 1023;
    float* base = out + ((size_t)(b0 + bl) * 3 + n) * 1024 + pix;
    float bb = bias[n];
    v4f o0, o1;
#pragma unroll
    for (int r = 0; r < 4; ++r) {
      float t0 = v[r] + bb, t1 = v[r + 4] + bb;
      o0[r] = 1.f / (1.f + __expf(-t0));
      o1[r] = 1.f / (1.f + __expf(-t1));
    }
    *(v4f*)base = o0; *(v4f*)(base + 4) = o1;
  }
};

// ---------------------------------------------------------------------------
// Weight repack kernels: W[n][k] f16 (B^T layout), K contiguous per column.
// ---------------------------------------------------------------------------
__global__ void pack_w1(const float* ew1, _Float16* W) {       // [128][64]
  for (int i = blockIdx.x * blockDim.x + threadIdx.x; i < 128 * 64;
       i += gridDim.x * blockDim.x) {
    int o = i >> 6, k = i & 63;
    float v = 0.f;
    if (k < 48) { int cin = k >> 4, tap = k & 15; v = ew1[(o * 3 + cin) * 16 + tap]; }
    W[i] = (_Float16)v;
  }
}
__global__ void pack_w2(const float* ew2, _Float16* W) {       // [128][2048]
  for (int i = blockIdx.x * blockDim.x + threadIdx.x; i < 128 * 2048;
       i += gridDim.x * blockDim.x) {
    int o = i >> 11, k = i & 2047, cin = k >> 4, tap = k & 15;
    W[i] = (_Float16)ew2[(o * 128 + cin) * 16 + tap];
  }
}
__global__ void pack_w3(const float* ew3, _Float16* W) {       // [64][1152]
  for (int i = blockIdx.x * blockDim.x + threadIdx.x; i < 64 * 1152;
       i += gridDim.x * blockDim.x) {
    int o = i / 1152, k = i - o * 1152, cin = k / 9, tap = k - cin * 9;
    W[i] = (_Float16)ew3[(o * 128 + cin) * 9 + tap];
  }
}
__global__ void pack_a1(const float* dw1, _Float16* A1) {      // [16][128][64]
  for (int i = blockIdx.x * blockDim.x + threadIdx.x; i < 16 * 128 * 64;
       i += gridDim.x * blockDim.x) {
    int type = i >> 13, rem = i & 8191, o = rem >> 6, cin = rem & 63;
    int rt = type >> 2, ct = type & 3;
    // valid-tap sets per row/col type: {1}, {1,3}, {0,2}, {2}
    const int T[4][2] = {{1, -1}, {1, 3}, {0, 2}, {2, -1}};
    float s = 0.f;
    for (int a = 0; a < 2; ++a) {
      int ky = T[rt][a]; if (ky < 0) continue;
      for (int c = 0; c < 2; ++c) {
        int kx = T[ct][c]; if (kx < 0) continue;
        s += dw1[(cin * 128 + o) * 16 + ky * 4 + kx];
      }
    }
    A1[i] = (_Float16)s;
  }
}
__global__ void pack_w2p(const float* dw2, _Float16* W) {      // [4][128][512]
  for (int i = blockIdx.x * blockDim.x + threadIdx.x; i < 4 * 128 * 512;
       i += gridDim.x * blockDim.x) {
    int pclass = i >> 16, rem = i & 65535, o = rem >> 9, k = rem & 511;
    int cin = k >> 2, tsy = (k >> 1) & 1, tsx = k & 1;
    int py = pclass >> 1, px = pclass & 1;
    int ky = (1 - py) + 2 * tsy, kx = (1 - px) + 2 * tsx;
    W[i] = (_Float16)dw2[(cin * 128 + o) * 16 + ky * 4 + kx];
  }
}
__global__ void pack_wd3(const float* dw3, _Float16* W) {      // [16][1152]
  for (int i = blockIdx.x * blockDim.x + threadIdx.x; i < 16 * 1152;
       i += gridDim.x * blockDim.x) {
    int o = i / 1152, k = i - o * 1152;
    float v = 0.f;
    if (o < 3) { int cin = k / 9, tap = k - cin * 9; v = dw3[(o * 128 + cin) * 9 + tap]; }
    W[i] = (_Float16)v;
  }
}

// ---------------------------------------------------------------------------
// Pool / VQ / loss
// ---------------------------------------------------------------------------
__global__ void pool_kernel(const float* c3, const float* eb3, float* z) {
  int g = blockIdx.x * blockDim.x + threadIdx.x;   // B*64
  if (g >= 1024 * 64) return;
  const float* s = c3 + (size_t)g * 64;
  float t = 0.f;
#pragma unroll 8
  for (int j = 0; j < 64; ++j) t += s[j];
  z[g] = t * (1.f / 64.f) + eb3[g & 63];
}

__global__ __launch_bounds__(256) void vq_kernel(const float* z, const float* cb,
                                                 float* qout, _Float16* qf16,
                                                 float* sqerr) {
  const int b = blockIdx.x, tid = threadIdx.x;
  __shared__ float zf[64];
  __shared__ float bd[256];
  __shared__ int   bi[256];
  __shared__ float se[64];
  if (tid < 64) zf[tid] = z[b * 64 + tid];
  __syncthreads();
  float best = 3.4e38f; int besti = 0x7fffffff;
  for (int c = tid; c < 512; c += 256) {
    const float* cv = cb + c * 64;
    float dot = 0.f, nrm = 0.f;
#pragma unroll 8
    for (int j = 0; j < 64; ++j) { float w = cv[j]; dot += w * zf[j]; nrm += w * w; }
    float d = nrm - 2.f * dot;                   // |z|^2 constant -> drop for argmin
    if (d < best || (d == best && c < besti)) { best = d; besti = c; }
  }
  bd[tid] = best; bi[tid] = besti;
  __syncthreads();
  for (int s = 128; s > 0; s >>= 1) {
    if (tid < s) {
      if (bd[tid + s] < bd[tid] || (bd[tid + s] == bd[tid] && bi[tid + s] < bi[tid])) {
        bd[tid] = bd[tid + s]; bi[tid] = bi[tid + s];
      }
    }
    __syncthreads();
  }
  int idx = bi[0];
  if (tid < 64) {
    float qv = cb[idx * 64 + tid];
    qout[b * 64 + tid] = qv;                     // q_st output (== q numerically)
    qf16[b * 64 + tid] = (_Float16)qv;
    float d = qv - zf[tid];
    se[tid] = d * d;
  }
  __syncthreads();
  if (tid == 0) {
    float s = 0.f;
    for (int j = 0; j < 64; ++j) s += se[j];
    sqerr[b] = s;
  }
}

__global__ void loss_kernel(const float* sq, float* out) {
  __shared__ float sh[256];
  int tid = threadIdx.x;
  float t = 0.f;
  for (int i = tid; i < 1024; i += 256) t += sq[i];
  sh[tid] = t; __syncthreads();
  for (int s = 128; s > 0; s >>= 1) { if (tid < s) sh[tid] += sh[tid + s]; __syncthreads(); }
  if (tid == 0) *out = 1.25f * sh[0] * (1.f / 65536.f);   // (1+COMMIT)*mean, B*EDIM=65536
}

// ---------------------------------------------------------------------------
// Host orchestration
// ---------------------------------------------------------------------------
extern "C" void kernel_launch(void* const* d_in, const int* in_sizes, int n_in,
                              void* d_out, int out_size, void* d_ws, size_t ws_size,
                              hipStream_t stream) {
  const float* x   = (const float*)d_in[0];
  const float* ew1 = (const float*)d_in[1];
  const float* eb1 = (const float*)d_in[2];
  const float* ew2 = (const float*)d_in[3];
  const float* eb2 = (const float*)d_in[4];
  const float* ew3 = (const float*)d_in[5];
  const float* eb3 = (const float*)d_in[6];
  const float* cb  = (const float*)d_in[7];
  const float* dw1 = (const float*)d_in[8];
  const float* db1 = (const float*)d_in[9];
  const float* dw2 = (const float*)d_in[10];
  const float* db2 = (const float*)d_in[11];
  const float* dw3 = (const float*)d_in[12];
  const float* db3 = (const float*)d_in[13];
  (void)in_sizes; (void)n_in; (void)ws_size;

  char* p = (char*)d_ws;
  auto carve = [&](size_t bytes) -> char* {
    char* r = p; p += (bytes + 255) & ~(size_t)255; return r;
  };
  _Float16* W1   = (_Float16*)carve(128 * 64 * 2);
  _Float16* Wc2  = (_Float16*)carve(128 * 2048 * 2);
  _Float16* Wc3  = (_Float16*)carve(64 * 1152 * 2);
  _Float16* A1   = (_Float16*)carve(16 * 128 * 64 * 2);
  _Float16* W2p  = (_Float16*)carve(4 * 128 * 512 * 2);
  _Float16* W3f  = (_Float16*)carve(16 * 1152 * 2);
  _Float16* h1e  = (_Float16*)carve((size_t)1024 * 128 * 256 * 2);  // 64 MB
  _Float16* h2e  = (_Float16*)carve((size_t)1024 * 128 * 64 * 2);   // 16 MB
  float*    c3o  = (float*)   carve((size_t)1024 * 64 * 64 * 4);    // 16 MB
  float*    z    = (float*)   carve(1024 * 64 * 4);
  _Float16* qf16 = (_Float16*)carve(1024 * 64 * 2);
  float*    sqer = (float*)   carve(1024 * 4);
  _Float16* h1t  = (_Float16*)carve((size_t)1024 * 16 * 128 * 2);   // 4 MB
  _Float16* h2d  = (_Float16*)carve((size_t)256 * 128 * 1024 * 2);  // 64 MB chunk

  float* recon = (float*)d_out;
  float* qout  = (float*)d_out + (size_t)1024 * 3 * 1024;
  float* lossp = (float*)d_out + (out_size - 1);

  // weight repacks
  pack_w1 <<<32,  256, 0, stream>>>(ew1, W1);
  pack_w2 <<<256, 256, 0, stream>>>(ew2, Wc2);
  pack_w3 <<<128, 256, 0, stream>>>(ew3, Wc3);
  pack_a1 <<<32,  256, 0, stream>>>(dw1, A1);
  pack_w2p<<<256, 256, 0, stream>>>(dw2, W2p);
  pack_wd3<<<32,  256, 0, stream>>>(dw3, W3f);

  // encoder
  MapConv1 m1{x, eb1, h1e};
  gemm_conv<64, 8, MapConv1><<<2048, 256, 0, stream>>>(m1, W1);     // M=262144
  MapConv2 m2{h1e, eb2, h2e};
  gemm_conv<2048, 8, MapConv2><<<512, 256, 0, stream>>>(m2, Wc2);   // M=65536
  MapConv3 m3{h2e, c3o};
  gemm_conv<1152, 4, MapConv3><<<512, 256, 0, stream>>>(m3, Wc3);   // M=65536
  pool_kernel<<<256, 256, 0, stream>>>(c3o, eb3, z);

  // vector quantizer + loss
  vq_kernel<<<1024, 256, 0, stream>>>(z, cb, qout, qf16, sqer);
  loss_kernel<<<1, 256, 0, stream>>>(sqer, lossp);

  // decoder: convT1 collapsed to 16 type GEMMs (input spatially constant)
  MapH1 mh{qf16, db1, h1t};
  gemm_conv<64, 8, MapH1><<<dim3(8, 16), 256, 0, stream>>>(mh, A1); // M=1024 x 16 types

  // decoder tail, chunked over batch to bound workspace
  for (int c = 0; c < 4; ++c) {
    int b0 = c * 256;
    MapT2 mt{h1t, db2, h2d, b0};
    gemm_conv<512, 8, MapT2><<<2048, 256, 0, stream>>>(mt, W2p);    // M=262144
    MapD3 md{h2d, db3, recon, b0};
    gemm_conv<1152, 1, MapD3><<<2048, 256, 0, stream>>>(md, W3f);   // M=262144
  }
}
